// ModifiedInnerShiftTriple_25864293056522
// MI455X (gfx1250) — compile-verified, matrix operand order checked by reading, not run on hardware
//
#include <hip/hip_runtime.h>
#include <stdint.h>

// ---------------------------------------------------------------------------
// ModifiedInnerShiftTriple for MI455X (gfx1250, wave32, WMMA).
//
// Shapes: x (2,128,96,96) f32, mask (96,96) i32.  B=2, c2=64, hw=9216.
// Fused bf16 WMMA GEMM (lat @ lat_n^T) + column-masked argmax; the 680MB sim
// matrix is never materialized.  All working data (<15MB) lives in L2.
// ---------------------------------------------------------------------------

typedef __attribute__((ext_vector_type(16))) __bf16 v16bf;
typedef __attribute__((ext_vector_type(8)))  float  v8f;

#define HW     9216      // 96*96
#define NTILES 576       // HW/16
#define BATCH  2

__device__ __forceinline__ unsigned short f32_to_bf16_rne(float f) {
  union { float f; unsigned int u; } x; x.f = f;
  unsigned int lsb = (x.u >> 16) & 1u;
  x.u += 0x7fffu + lsb;
  return (unsigned short)(x.u >> 16);
}

// Fragment layout: [b][tile][khalf][lane][16 elems] -- each lane's v16bf is a
// contiguous, 32B-aligned run (two 128-bit loads per lane per fragment).
__device__ __forceinline__ size_t packIdx(int b, int tile, int khalf, int lane) {
  return ((((size_t)b * NTILES + tile) * 2 + khalf) * 32 + lane) * 16;
}

// ---------------------------------------------------------------------------
// Kernel 1: copy former|latter (x channels 0..127) into out channels 0..127.
// float4-vectorized; 589,824 float4 elements.
// ---------------------------------------------------------------------------
__global__ void __launch_bounds__(256)
copy_former_latter(const float4* __restrict__ x4, float4* __restrict__ out4) {
  const int QPC = HW / 4;  // 2304 float4 per channel
  int id = blockIdx.x * blockDim.x + threadIdx.x;
  int b  = id / (128 * QPC);
  int r  = id - b * (128 * QPC);
  int ch = r / QPC;
  int q  = r - ch * QPC;
  out4[((size_t)b * 192 + ch) * QPC + q] = x4[id];
}

// ---------------------------------------------------------------------------
// Kernel 2: per-pixel L2 norm of latter + bf16 packing of A (lat) and
// B (lat_n) WMMA fragments.  One thread per (b, pixel): 18,432 threads.
// ---------------------------------------------------------------------------
__global__ void __launch_bounds__(256)
norm_pack(const float* __restrict__ x,
          unsigned short* __restrict__ apack,
          unsigned short* __restrict__ bpack) {
  int id = blockIdx.x * blockDim.x + threadIdx.x;   // 0 .. 2*HW-1
  int b  = id / HW;
  int p  = id - b * HW;

  const float* lat = x + ((size_t)b * 128 + 64) * HW + p;  // latter, ch-stride HW
  float v[64];
  float ss = 0.f;
#pragma unroll
  for (int c = 0; c < 64; ++c) {
    float t = lat[(size_t)c * HW];
    v[c] = t;
    ss += t * t;
  }
  float inv = 1.f / fmaxf(sqrtf(ss), 1e-8f);

  int tile = p >> 4;
  int m    = p & 15;   // row within tile (A) == column within tile (B)
#pragma unroll
  for (int kh = 0; kh < 2; ++kh) {
#pragma unroll
    for (int lh = 0; lh < 2; ++lh) {
      unsigned short* ao = apack + packIdx(b, tile, kh, m + lh * 16);
      unsigned short* bo = bpack + packIdx(b, tile, kh, m + lh * 16);
#pragma unroll
      for (int e = 0; e < 16; ++e) {
        int K = kh * 32 + lh * 16 + e;
        ao[e] = f32_to_bf16_rne(v[K]);
        bo[e] = f32_to_bf16_rne(v[K] * inv);
      }
    }
  }
}

// ---------------------------------------------------------------------------
// Kernel 3: fused WMMA GEMM + masked argmax + gather/shift write.
// One wave32 per 16-row tile; 8 waves per block; 2*576 = 1152 wave-tasks.
// ---------------------------------------------------------------------------
__global__ void __launch_bounds__(256)
sim_argmax_shift(const float* __restrict__ x,
                 const int* __restrict__ mask,
                 const unsigned short* __restrict__ apack,
                 const unsigned short* __restrict__ bpack,
                 float* __restrict__ out) {
  __shared__ unsigned char s_flag[HW];   // 9KB flag table (shared by all waves)
  __shared__ int s_idx[8][16];           // per-wave argmax row indices

  for (int p = threadIdx.x; p < HW; p += blockDim.x)
    s_flag[p] = (mask[p] >= 1) ? 1 : 0;
  __syncthreads();

  int wave = threadIdx.x >> 5;
  int lane = threadIdx.x & 31;
  int wid  = blockIdx.x * 8 + wave;      // 0 .. 1151
  int b    = wid / NTILES;
  int it   = wid - b * NTILES;

  // A fragments for this wave's 16 rows (K = 0..31, 32..63), kept in VGPRs.
  const v16bf a0 = *reinterpret_cast<const v16bf*>(apack + packIdx(b, it, 0, lane));
  const v16bf a1 = *reinterpret_cast<const v16bf*>(apack + packIdx(b, it, 1, lane));

  float best[8];
  int   bidx[8];
#pragma unroll
  for (int r = 0; r < 8; ++r) { best[r] = -1e30f; bidx[r] = 0; }

  const int mycol = lane & 15;           // C/D layout: lane -> column N
  for (int jt = 0; jt < NTILES; ++jt) {
    const v16bf b0 = *reinterpret_cast<const v16bf*>(bpack + packIdx(b, jt, 0, lane));
    const v16bf b1 = *reinterpret_cast<const v16bf*>(bpack + packIdx(b, jt, 1, lane));
    if (jt + 8 < NTILES)                 // stream B fragments ahead (global_prefetch_b8)
      __builtin_prefetch(bpack + packIdx(b, jt + 8, 0, lane), 0, 3);

    v8f c = {};
    c = __builtin_amdgcn_wmma_f32_16x16x32_bf16(false, a0, false, b0,
                                                (short)0, c, false, false);
    c = __builtin_amdgcn_wmma_f32_16x16x32_bf16(false, a1, false, b1,
                                                (short)0, c, false, false);

    int jcol = jt * 16 + mycol;
    if (!s_flag[jcol]) {                 // masked columns contribute NEG -> skip
#pragma unroll
      for (int r = 0; r < 8; ++r) {
        float vv = c[r];
        if (vv > best[r]) { best[r] = vv; bidx[r] = jcol; }
      }
    }
  }

  // Argmax reduction across each 16-lane half (rows r and 8+r independently).
#pragma unroll
  for (int r = 0; r < 8; ++r) {
    float bv = best[r];
    int   bi = bidx[r];
    for (int off = 8; off > 0; off >>= 1) {
      float ov = __shfl_xor(bv, off, 16);
      int   oi = __shfl_xor(bi, off, 16);
      if (ov > bv || (ov == bv && oi < bi)) { bv = ov; bi = oi; }
    }
    if ((lane & 15) == 0) {
      int m = (lane >> 4) * 8 + r;       // lane0 -> row r, lane16 -> row 8+r
      s_idx[wave][m] = bi;
    }
  }
  __syncthreads();

  // shift[b, c, p] = flag[p] ? former[b, c, idx[p]] : 0   (out channels 128..191)
  const float* fmr   = x   + (size_t)b * 128 * HW;
  float*       shift = out + ((size_t)b * 192 + 128) * HW;
  const int i0 = it * 16;
  for (int t = lane; t < 16 * 64; t += 32) {
    int m  = t & 15;
    int ch = t >> 4;
    int p  = i0 + m;
    float val = 0.f;
    if (s_flag[p]) val = fmr[(size_t)ch * HW + s_idx[wave][m]];
    shift[(size_t)ch * HW + p] = val;
  }
}

// ---------------------------------------------------------------------------
extern "C" void kernel_launch(void* const* d_in, const int* in_sizes, int n_in,
                              void* d_out, int out_size, void* d_ws, size_t ws_size,
                              hipStream_t stream) {
  (void)in_sizes; (void)n_in; (void)out_size; (void)ws_size;
  const float* x    = (const float*)d_in[0];   // (2,128,96,96) f32
  const int*   mask = (const int*)d_in[1];     // (96,96) i32
  float*       out  = (float*)d_out;           // (2,192,96,96) f32

  // Workspace: A-pack and B-pack, each 2*576*2*32*16 bf16 = 2.25 MiB.
  unsigned short* apack = (unsigned short*)d_ws;
  unsigned short* bpack = apack + (size_t)BATCH * NTILES * 2 * 32 * 16;

  copy_former_latter<<<(BATCH * 128 * (HW / 4)) / 256, 256, 0, stream>>>(
      (const float4*)x, (float4*)out);
  norm_pack<<<(BATCH * HW) / 256, 256, 0, stream>>>(x, apack, bpack);
  sim_argmax_shift<<<(BATCH * NTILES) / 8, 256, 0, stream>>>(
      x, mask, apack, bpack, out);
}
